// BiGraphSAGEDecoder_18167711662116
// MI455X (gfx1250) — matrix operand order repaired and checked by reading, not compiled, not run on hardware
//
#include <hip/hip_runtime.h>

// ---------------------------------------------------------------------------
// BiGraphSAGE decoder for MI455X (gfx1250).
// bf16 WMMA (v_wmma_f32_16x16x32_bf16) with fp32 accumulate for all large
// GEMMs; fp32 epilogues.  GEMM uses LDS double-buffering with
// global_load_async_to_lds_b128 for the A tile (ASYNCcnt-tracked DMA into
// LDS) overlapped with WMMA compute on the previous tile.
// ---------------------------------------------------------------------------

typedef __attribute__((ext_vector_type(16))) __bf16 v16bf;
typedef __attribute__((ext_vector_type(8)))  __bf16 bf16x8;
typedef __attribute__((ext_vector_type(8)))  float  v8f;

#define GN   2048            // N (graph nodes)
#define GB   4               // batch
#define CATD 768             // concat feature dim (3*256)

__device__ __forceinline__ __bf16 f2bf(float f) {
  // round-to-nearest-even fp32 -> bf16 (bit trick; NaN payload not preserved)
  unsigned u = __float_as_uint(f);
  unsigned r = (u + 0x7FFFu + ((u >> 16) & 1u)) >> 16;
  unsigned short s = (unsigned short)r;
  __bf16 b;
  __builtin_memcpy(&b, &s, 2);
  return b;
}

__device__ __forceinline__ v16bf ld_frag(const __bf16* lo, const __bf16* hi) {
  v16bf r;
  ((bf16x8*)&r)[0] = *(const bf16x8*)lo;
  ((bf16x8*)&r)[1] = *(const bf16x8*)hi;
  return r;
}

// ---------------------------------------------------------------------------
// elementwise fp32 -> bf16 convert
// ---------------------------------------------------------------------------
__global__ void conv_bf_kernel(const float* __restrict__ src,
                               __bf16* __restrict__ dst, long long n) {
  long long i = (long long)blockIdx.x * 256 + threadIdx.x;
  if (i < n) dst[i] = f2bf(src[i]);
}

// ---------------------------------------------------------------------------
// masked adjacency prep (optionally transposed), fp32 -> bf16
//   dst[b][i][j] = adj[b][r][c] * (mask(r,c) ? w[r][c] : 1),  (r,c) = T?(j,i):(i,j)
//   mask(r,c) = (r < N-2 && c < N-2)
// ---------------------------------------------------------------------------
__global__ void prep_adj_kernel(const float* __restrict__ adj,
                                const float* __restrict__ w,
                                __bf16* __restrict__ dst, int transpose) {
  const long long b = blockIdx.z;
  long long idx = (long long)blockIdx.x * 256 + threadIdx.x;   // < N*N
  int i = (int)(idx >> 11);
  int j = (int)(idx & (GN - 1));
  int r = transpose ? j : i;
  int c = transpose ? i : j;
  float m = (r < GN - 2 && c < GN - 2) ? w[(long long)r * GN + c] : 1.0f;
  float v = adj[b * (long long)GN * GN + (long long)r * GN + c] * m;
  dst[b * (long long)GN * GN + idx] = f2bf(v);
}

// ---------------------------------------------------------------------------
// bf16 GEMM, fp32 accumulate:  C[bz] = A[bz] (MxK, row-major) * B[bz] (KxN)
// Block = 256 threads (8 wave32).  Tile: 128(M) x 64(N), K-step 32.
// LDS double-buffered: A tile staged with global_load_async_to_lds_b128
// (ASYNCcnt), B tile transposed through VGPRs into LDS.  Each wave computes a
// 32x32 macro tile = 2x2 WMMA fragments.
// Output: fp32 (Cf) or bf16 (Cb) selected by non-null pointer.
// Requires: M%128==0, N%64==0, K%32==0, 16B-aligned operands, lda/ldb%8==0.
// ---------------------------------------------------------------------------
__global__ __launch_bounds__(256) void gemm_bf16_kernel(
    const __bf16* __restrict__ A, int lda, long long sA,
    const __bf16* __restrict__ B, int ldb, long long sB,
    float* __restrict__ Cf, __bf16* __restrict__ Cb, int ldc, long long sC,
    int K) {
  __shared__ __bf16 As[2][128 * 32];   // A tiles, row-major (row stride 32)
  __shared__ __bf16 Bts[2][64 * 32];   // B tiles transposed: [n][k]

  const int tid  = threadIdx.x;
  const int wv   = tid >> 5;
  const int lane = tid & 31;
  const int wr   = (wv >> 1) * 32;      // wave row base within 128
  const int wc   = (wv & 1) * 32;       // wave col base within 64
  const int lrow = lane & 15;
  const int hk   = lane >> 4;           // lane-half select
  const int k8   = hk * 8;              // A: K half offset within 16
  const int k16  = hk * 16;             // unused; B fragments also use k8

  const long long m0 = (long long)blockIdx.y * 128;
  const long long n0 = (long long)blockIdx.x * 64;

  const __bf16* Ab = A + (long long)blockIdx.z * sA;
  const __bf16* Bb = B + (long long)blockIdx.z * sB;

  // LDS byte offsets of the two A buffers (low 32 bits of the flat address
  // are the in-aperture LDS offset on gfx1250).
  const unsigned asbase0 = (unsigned)(unsigned long long)(const void*)&As[0][0];
  const unsigned asbase1 = (unsigned)(unsigned long long)(const void*)&As[1][0];

  // per-thread staging geometry
  const int arow0 = tid >> 2;                  // A chunk 0 row   (0..63)
  const int akc   = (tid & 3) << 3;            // A chunk K offset (0/8/16/24)
  const int bkr   = tid >> 3;                  // B row (K)        (0..31)
  const int bnc   = (tid & 7) << 3;            // B col base       (0..56)

  // stage tile at k0 into buffer `buf`
  auto stage = [&](int k0, int buf) {
    const unsigned asb = buf ? asbase1 : asbase0;
    // ---- A tile 128x32: two async 16B DMA copies per thread ----
#pragma unroll
    for (int t = 0; t < 2; ++t) {
      int row = arow0 + t * 64;
      const __bf16* gsrc = Ab + (m0 + row) * (long long)lda + k0 + akc;
      unsigned ldst = asb + (unsigned)(row * 32 + akc) * 2u;
      asm volatile("global_load_async_to_lds_b128 %0, %1, off"
                   :: "v"(ldst), "v"((unsigned long long)gsrc)
                   : "memory");
    }
    // ---- B tile 32x64: load 16B, scatter transposed into LDS ----
    bf16x8 v = *(const bf16x8*)(Bb + (long long)(k0 + bkr) * ldb + n0 + bnc);
    __bf16* bt = &Bts[buf][0];
#pragma unroll
    for (int i = 0; i < 8; ++i) bt[(bnc + i) * 32 + bkr] = v[i];
  };

  v8f acc00 = {}, acc01 = {}, acc10 = {}, acc11 = {};

  // prologue: stage tile 0
  stage(0, 0);
  asm volatile("s_wait_asynccnt 0x0" ::: "memory");
  __syncthreads();

  int cur = 0;
  for (int k0 = 0; k0 < K; k0 += 32) {
    // prefetch next tile into the other buffer while computing this one
    if (k0 + 32 < K) stage(k0 + 32, cur ^ 1);

    const __bf16* as = &As[cur][0];
    const __bf16* bs = &Bts[cur][0];

    v16bf a0 = ld_frag(&as[(wr + lrow) * 32 + k8],
                       &as[(wr + lrow) * 32 + 16 + k8]);
    v16bf a1 = ld_frag(&as[(wr + 16 + lrow) * 32 + k8],
                       &as[(wr + 16 + lrow) * 32 + 16 + k8]);
    v16bf b0 = ld_frag(&bs[(wc + lrow) * 32 + k8],
                       &bs[(wc + lrow) * 32 + 16 + k8]);
    v16bf b1 = ld_frag(&bs[(wc + 16 + lrow) * 32 + k8],
                       &bs[(wc + 16 + lrow) * 32 + 16 + k8]);

    acc00 = __builtin_amdgcn_wmma_f32_16x16x32_bf16(false, a0, false, b0,
                                                    (short)0, acc00, false, false);
    acc01 = __builtin_amdgcn_wmma_f32_16x16x32_bf16(false, a0, false, b1,
                                                    (short)0, acc01, false, false);
    acc10 = __builtin_amdgcn_wmma_f32_16x16x32_bf16(false, a1, false, b0,
                                                    (short)0, acc10, false, false);
    acc11 = __builtin_amdgcn_wmma_f32_16x16x32_bf16(false, a1, false, b1,
                                                    (short)0, acc11, false, false);

    // make the prefetch (async A DMA + transposed B stores) visible, then
    // rendezvous before flipping buffers
    asm volatile("s_wait_asynccnt 0x0" ::: "memory");
    __syncthreads();
    cur ^= 1;
  }

  // ---- store: C/D layout: VGPR v -> M = v + 8*(lane>=16), N = lane&15 ----
  const int rsel = hk * 8;
  const int cl   = lane & 15;
  float*  Cfb = Cf ? Cf + (long long)blockIdx.z * sC : (float*)0;
  __bf16* Cbb = Cb ? Cb + (long long)blockIdx.z * sC : (__bf16*)0;

  v8f accs[4] = {acc00, acc01, acc10, acc11};
#pragma unroll
  for (int t = 0; t < 4; ++t) {
    int ti = t >> 1, tj = t & 1;
    long long rb = m0 + wr + ti * 16 + rsel;
    long long cb = n0 + wc + tj * 16 + cl;
#pragma unroll
    for (int v = 0; v < 8; ++v) {
      long long off = (rb + v) * (long long)ldc + cb;
      if (Cbb) Cbb[off] = f2bf(accs[t][v]);
      else     Cfb[off] = accs[t][v];
    }
  }
}

// ---------------------------------------------------------------------------
// per-row L2 normalize (768 wide) + leaky ReLU; dual write fp32 + bf16
// ---------------------------------------------------------------------------
__global__ __launch_bounds__(256) void norm_leaky_kernel(
    const float* __restrict__ cat, float* __restrict__ hf,
    __bf16* __restrict__ hb) {
  const long long row = blockIdx.x;              // 0 .. B*N-1
  const float* src = cat + row * CATD;
  const int tid = threadIdx.x;                   // 256

  float a = src[tid];
  float b = src[tid + 256];
  float c = src[tid + 512];

  __shared__ float red[256];
  red[tid] = a * a + b * b + c * c;
  __syncthreads();
#pragma unroll
  for (int s = 128; s > 0; s >>= 1) {
    if (tid < s) red[tid] += red[tid + s];
    __syncthreads();
  }
  float inv = 1.0f / fmaxf(sqrtf(red[0]), 1e-12f);

  float ya = a * inv; ya = (ya >= 0.0f) ? ya : 0.1f * ya;
  float yb = b * inv; yb = (yb >= 0.0f) ? yb : 0.1f * yb;
  float yc = c * inv; yc = (yc >= 0.0f) ? yc : 0.1f * yc;

  long long base = row * CATD;
  hf[base + tid]       = ya;  hb[base + tid]       = f2bf(ya);
  hf[base + tid + 256] = yb;  hb[base + tid + 256] = f2bf(yb);
  hf[base + tid + 512] = yc;  hb[base + tid + 512] = f2bf(yc);
}

// ---------------------------------------------------------------------------
// final decode: ypred[b] = sum(((a@P1)@P2) * (b@P1));  tiny -> one block
// ---------------------------------------------------------------------------
__global__ __launch_bounds__(128) void final_decode_kernel(
    const float* __restrict__ h, const float* __restrict__ p1,
    const float* __restrict__ p2, float* __restrict__ out) {
  __shared__ float sh[128];
  __shared__ float red[128];
  const int j = threadIdx.x;  // 128
  for (int b = 0; b < GB; ++b) {
    const float* da = h + ((long long)b * GN + (GN - 2)) * CATD;
    const float* db = h + ((long long)b * GN + (GN - 1)) * CATD;
    float va = 0.0f, vb = 0.0f;
    for (int k = 0; k < CATD; ++k) {
      float p = p1[(long long)k * 128 + j];
      va += da[k] * p;
      vb += db[k] * p;
    }
    sh[j] = va;
    __syncthreads();
    float tj = 0.0f;
    for (int i = 0; i < 128; ++i) tj += sh[i] * p2[i * 128 + j];
    red[j] = tj * vb;
    __syncthreads();
#pragma unroll
    for (int s = 64; s > 0; s >>= 1) {
      if (j < s) red[j] += red[j + s];
      __syncthreads();
    }
    if (j == 0) out[b] = red[0];
    __syncthreads();
  }
}

// ---------------------------------------------------------------------------
// host orchestration
// ---------------------------------------------------------------------------
extern "C" void kernel_launch(void* const* d_in, const int* in_sizes, int n_in,
                              void* d_out, int out_size, void* d_ws,
                              size_t ws_size, hipStream_t stream) {
  (void)in_sizes; (void)n_in; (void)out_size; (void)ws_size;

  const float* x     = (const float*)d_in[0];   // (B,N,256)
  const float* adj   = (const float*)d_in[1];   // (B,N,N)
  const float* upinv = (const float*)d_in[2];   // (B,N,N)
  const float* dninv = (const float*)d_in[3];   // (B,N,N)
  const float* p1    = (const float*)d_in[19];  // (768,128)
  const float* p2    = (const float*)d_in[20];  // (128,128)

  const long long NNe = (long long)GN * GN;           // 4 Mi elems
  const long long HNe = (long long)GN * CATD;         // 1.5 Mi elems

  char* cur = (char*)d_ws;
  auto carve = [&](size_t bytes) -> char* {
    char* p = cur;
    cur += (bytes + 255) & ~(size_t)255;
    return p;
  };

  __bf16* upinv_bf = (__bf16*)carve((size_t)GB * NNe * 2);
  __bf16* dninv_bf = (__bf16*)carve((size_t)GB * NNe * 2);
  __bf16* upadj_bf = (__bf16*)carve((size_t)GB * NNe * 2);
  __bf16* dnadj_bf = (__bf16*)carve((size_t)GB * NNe * 2);
  __bf16* h_bf     = (__bf16*)carve((size_t)GB * HNe * 2);
  __bf16* tmp_bf   = (__bf16*)carve((size_t)GB * HNe * 2);
  __bf16* upx_bf   = (__bf16*)carve((size_t)GB * HNe * 2);
  __bf16* dnx_bf   = (__bf16*)carve((size_t)GB * HNe * 2);
  float*  cat_f32  = (float*)carve((size_t)GB * HNe * 4);
  float*  h_f32    = (float*)carve((size_t)GB * HNe * 4);
  __bf16* wup_bf   = (__bf16*)carve((size_t)CATD * 256 * 2);
  __bf16* wdn_bf   = (__bf16*)carve((size_t)CATD * 256 * 2);
  __bf16* wbs_bf   = (__bf16*)carve((size_t)CATD * 256 * 2);

  auto cgrid = [](long long n) { return dim3((unsigned)((n + 255) / 256)); };

  // one-time converts
  conv_bf_kernel<<<cgrid(GB * NNe), 256, 0, stream>>>(upinv, upinv_bf, GB * NNe);
  conv_bf_kernel<<<cgrid(GB * NNe), 256, 0, stream>>>(dninv, dninv_bf, GB * NNe);
  conv_bf_kernel<<<cgrid((long long)GB * GN * 256), 256, 0, stream>>>(
      x, h_bf, (long long)GB * GN * 256);

  auto gemm = [&](const __bf16* A, int lda, long long sA, const __bf16* B,
                  int ldb, long long sB, float* Cf, __bf16* Cb, int ldc,
                  long long sC, int M, int Nn, int K) {
    dim3 g((unsigned)(Nn / 64), (unsigned)(M / 128), GB);
    gemm_bf16_kernel<<<g, 256, 0, stream>>>(A, lda, sA, B, ldb, sB, Cf, Cb,
                                            ldc, sC, K);
  };

  const int dins[3] = {256, CATD, CATD};
  for (int l = 0; l < 3; ++l) {
    const float* up_w  = (const float*)d_in[4 + 5 * l + 0];
    const float* dn_w  = (const float*)d_in[4 + 5 * l + 1];
    const float* bias  = (const float*)d_in[4 + 5 * l + 2];
    const float* up_aw = (const float*)d_in[4 + 5 * l + 3];
    const float* dn_aw = (const float*)d_in[4 + 5 * l + 4];
    const int din = dins[l];
    const long long hstr = (long long)GN * din;

    // masked/weighted adjacency (up path pre-transposed), bf16
    dim3 gp((unsigned)(NNe / 256), 1, GB);
    prep_adj_kernel<<<gp, 256, 0, stream>>>(adj, up_aw, upadj_bf, 1);
    prep_adj_kernel<<<gp, 256, 0, stream>>>(adj, dn_aw, dnadj_bf, 0);

    // layer weights -> bf16
    conv_bf_kernel<<<cgrid((long long)din * 256), 256, 0, stream>>>(
        up_w, wup_bf, (long long)din * 256);
    conv_bf_kernel<<<cgrid((long long)din * 256), 256, 0, stream>>>(
        dn_w, wdn_bf, (long long)din * 256);
    conv_bf_kernel<<<cgrid((long long)din * 256), 256, 0, stream>>>(
        bias, wbs_bf, (long long)din * 256);

    // up_x = up_inv @ (up_adj^T @ h)
    gemm(upadj_bf, GN, NNe, h_bf, din, hstr, 0, tmp_bf, din, hstr, GN, din, GN);
    gemm(upinv_bf, GN, NNe, tmp_bf, din, hstr, 0, upx_bf, din, hstr, GN, din, GN);
    // down_x = down_inv @ (down_adj @ h)
    gemm(dnadj_bf, GN, NNe, h_bf, din, hstr, 0, tmp_bf, din, hstr, GN, din, GN);
    gemm(dninv_bf, GN, NNe, tmp_bf, din, hstr, 0, dnx_bf, din, hstr, GN, din, GN);

    // cat = [up_x@up_w | down_x@down_w | h@bias]  (fp32, ldc = 768)
    const long long cstr = (long long)GN * CATD;
    gemm(upx_bf, din, hstr, wup_bf, 256, 0, cat_f32 + 0,   0, CATD, cstr, GN, 256, din);
    gemm(dnx_bf, din, hstr, wdn_bf, 256, 0, cat_f32 + 256, 0, CATD, cstr, GN, 256, din);
    gemm(h_bf,   din, hstr, wbs_bf, 256, 0, cat_f32 + 512, 0, CATD, cstr, GN, 256, din);

    // normalize + leaky -> next layer h (fp32 + bf16)
    norm_leaky_kernel<<<dim3((unsigned)(GB * GN)), 256, 0, stream>>>(
        cat_f32, h_f32, h_bf);
  }

  final_decode_kernel<<<1, 128, 0, stream>>>(h_f32, p1, p2, (float*)d_out);
}
